// GPT2AttentionOri_56994216018544
// MI455X (gfx1250) — matrix operand order repaired and checked by reading, not compiled, not run on hardware
//
#include <hip/hip_runtime.h>

// B=2, S=2048, E=1024, H=16, HD=64
#define B_   2
#define S_   2048
#define E_   1024
#define H_   16
#define HD_  64
#define N3E  3072
#define MTOT 4096   // B*S

typedef __attribute__((ext_vector_type(16))) __bf16 v16bf;
typedef __attribute__((ext_vector_type(8)))  __bf16 v8bf;
typedef __attribute__((ext_vector_type(4)))  __bf16 v4bf;
typedef __attribute__((ext_vector_type(8)))  float  v8f;

// Build a 16-element bf16 fragment from two contiguous 8-element (16B) chunks.
__device__ __forceinline__ v16bf make_frag(const __bf16* lo, const __bf16* hi) {
  v8bf a = *(const v8bf*)lo;
  v8bf b = *(const v8bf*)hi;
  return __builtin_shufflevector(a, b, 0,1,2,3,4,5,6,7,8,9,10,11,12,13,14,15);
}

__device__ __forceinline__ v8f wmma_bf16(v16bf a, v16bf b, v8f c) {
  return __builtin_amdgcn_wmma_f32_16x16x32_bf16(false, a, false, b, (short)0, c,
                                                 false, false);
}

// ---------------------------------------------------------------------------
// Kernel 1: QKV = X @ W_attn + b_attn ; scatter to bf16 ws [seg][b][h][s][d]
// Grid (32, 24), block 256 (8 waves, 2x4), tile 128x128, K-step 64.
// ---------------------------------------------------------------------------
__global__ __launch_bounds__(256, 1)
void qkv_gemm_kernel(const float* __restrict__ X, const float* __restrict__ W,
                     const float* __restrict__ bias,
                     __bf16* __restrict__ qkv) {
  __shared__ __bf16 lds_a[128 * 72];   // [row][k] bf16, pitch 72
  __shared__ __bf16 lds_b[128 * 72];   // [n][k]   bf16 (transposed W)

  const int t     = threadIdx.x;
  const int lane  = t & 31;
  const int wave  = t >> 5;       // 0..7
  const int waveM = wave >> 2;    // 0..1
  const int waveN = wave & 3;     // 0..3
  const int m0 = blockIdx.x * 128;
  const int n0 = blockIdx.y * 128;
  const int hlf = (lane >> 4) & 1;
  const int l15 = lane & 15;

  v8f acc[4][2];
  #pragma unroll
  for (int mi = 0; mi < 4; ++mi)
    #pragma unroll
    for (int ni = 0; ni < 2; ++ni)
      acc[mi][ni] = (v8f){0.f,0.f,0.f,0.f,0.f,0.f,0.f,0.f};

  for (int k = 0; k < E_; k += 64) {
    // Stage A: 128x64 f32 -> bf16 LDS (packed 4-wide stores)
    #pragma unroll
    for (int j = 0; j < 8; ++j) {
      int flat = t + j * 256;             // 2048 float4
      int row  = flat >> 4;               // 16 float4 per row
      int col  = (flat & 15) * 4;
      const float* gp = X + (size_t)(m0 + row) * E_ + k + col;
      const float4 v = *(const float4*)gp;
      if (k + 64 < E_) __builtin_prefetch(gp + 64, 0, 3);
      v4bf o = {(__bf16)v.x, (__bf16)v.y, (__bf16)v.z, (__bf16)v.w};
      *(v4bf*)&lds_a[row * 72 + col] = o;
    }
    // Stage B transposed: W[k..k+63][n0..n0+127] -> lds_b[n][krow]
    #pragma unroll
    for (int j = 0; j < 8; ++j) {
      int flat = t + j * 256;             // 2048 float4
      int krow = flat >> 5;               // 32 float4 per row
      int ncol = (flat & 31) * 4;
      const float* gp = W + (size_t)(k + krow) * N3E + n0 + ncol;
      const float4 v = *(const float4*)gp;
      if (k + 64 < E_) __builtin_prefetch(gp + (size_t)64 * N3E, 0, 3);
      lds_b[(ncol + 0) * 72 + krow] = (__bf16)v.x;
      lds_b[(ncol + 1) * 72 + krow] = (__bf16)v.y;
      lds_b[(ncol + 2) * 72 + krow] = (__bf16)v.z;
      lds_b[(ncol + 3) * 72 + krow] = (__bf16)v.w;
    }
    __syncthreads();

    #pragma unroll
    for (int kc = 0; kc < 2; ++kc) {
      v16bf afrag[4];
      #pragma unroll
      for (int mi = 0; mi < 4; ++mi) {
        int row = waveM * 64 + mi * 16 + l15;
        const __bf16* p = lds_a + row * 72 + kc * 32 + hlf * 8;
        afrag[mi] = make_frag(p, p + 16);
      }
      v16bf bfrag[2];
      #pragma unroll
      for (int ni = 0; ni < 2; ++ni) {
        int n = waveN * 32 + ni * 16 + l15;
        const __bf16* p = lds_b + n * 72 + kc * 32 + hlf * 8;
        bfrag[ni] = make_frag(p, p + 16);
      }
      #pragma unroll
      for (int mi = 0; mi < 4; ++mi)
        #pragma unroll
        for (int ni = 0; ni < 2; ++ni)
          acc[mi][ni] = wmma_bf16(afrag[mi], bfrag[ni], acc[mi][ni]);
    }
    __syncthreads();
  }

  // Epilogue: bias + scatter to per-head QKV workspace (bf16)
  #pragma unroll
  for (int mi = 0; mi < 4; ++mi) {
    #pragma unroll
    for (int ni = 0; ni < 2; ++ni) {
      int n  = n0 + waveN * 32 + ni * 16 + l15;
      float bv = bias[n];
      int seg = n >> 10; int e = n & 1023; int h = e >> 6; int d = e & 63;
      #pragma unroll
      for (int r = 0; r < 8; ++r) {
        int m = m0 + waveM * 64 + mi * 16 + r + hlf * 8;
        int bb = m >> 11; int s = m & 2047;
        size_t dst = (((size_t)(seg * B_ + bb) * H_ + h) * S_ + s) * HD_ + d;
        qkv[dst] = (__bf16)(acc[mi][ni][r] + bv);
      }
    }
  }
}

// ---------------------------------------------------------------------------
// Kernel 2: causal flash attention per (b,h,64-query block).
// Grid (32 qblocks, 32 bh), block 128 (4 waves x 16 query rows each).
// ---------------------------------------------------------------------------
__global__ __launch_bounds__(128, 1)
void attn_kernel(const __bf16* __restrict__ qkv,
                 __bf16* __restrict__ attn_out) {
  __shared__ __bf16 lds_vt[64 * 72];      // V^T: [d][key], pitch 72
  __shared__ __bf16 lds_p[4][16 * 72];    // per-wave P: [row][key]

  const int t    = threadIdx.x;
  const int lane = t & 31;
  const int wave = t >> 5;                 // 0..3
  const int qbi  = blockIdx.x;             // query block (64 rows)
  const int bh   = blockIdx.y;
  const int b = bh >> 4, h = bh & 15;
  const int hlf = (lane >> 4) & 1;
  const int l15 = lane & 15;

  const size_t plane = (size_t)S_ * HD_;
  const __bf16* Qb = qkv + ((size_t)(0 * B_ + b) * H_ + h) * plane;
  const __bf16* Kb = qkv + ((size_t)(1 * B_ + b) * H_ + h) * plane;
  const __bf16* Vb = qkv + ((size_t)(2 * B_ + b) * H_ + h) * plane;

  const int q0 = qbi * 64 + wave * 16;     // this wave's first query row

  // Q A-fragments (HD=64 -> two K=32 chunks), contiguous global b128 loads
  v16bf qf[2];
  {
    const __bf16* qrow = Qb + (size_t)(q0 + l15) * HD_;
    #pragma unroll
    for (int c = 0; c < 2; ++c) {
      const __bf16* p = qrow + c * 32 + hlf * 8;
      qf[c] = make_frag(p, p + 16);
    }
  }

  v8f o[4];
  #pragma unroll
  for (int n = 0; n < 4; ++n) o[n] = (v8f){0.f,0.f,0.f,0.f,0.f,0.f,0.f,0.f};
  float mrow[8], lrow[8];
  #pragma unroll
  for (int r = 0; r < 8; ++r) { mrow[r] = -1e30f; lrow[r] = 0.f; }

  for (int kb = 0; kb <= qbi; ++kb) {
    __syncthreads();   // previous block's LDS consumers done
    // Stage V^T (64 keys x 64 dims) cooperatively
    #pragma unroll
    for (int i = 0; i < 4; ++i) {
      int flat = t + i * 128;              // 512 x 8-elt chunks
      int key  = flat >> 3;
      int d0   = (flat & 7) * 8;
      const v8bf v = *(const v8bf*)(Vb + (size_t)(kb * 64 + key) * HD_ + d0);
      #pragma unroll
      for (int q = 0; q < 8; ++q) lds_vt[(d0 + q) * 72 + key] = v[q];
    }
    __syncthreads();

    // Scores: Q(16x64) @ K^T(64x64) -> 4 tiles of 16x16 (scaled + masked)
    float sb[4][8];
    #pragma unroll
    for (int tt = 0; tt < 4; ++tt) {
      int key = kb * 64 + tt * 16 + l15;
      const __bf16* krow = Kb + (size_t)key * HD_;
      v8f s = (v8f){0.f,0.f,0.f,0.f,0.f,0.f,0.f,0.f};
      #pragma unroll
      for (int c = 0; c < 2; ++c) {
        const __bf16* p = krow + c * 32 + hlf * 8;
        v16bf kf = make_frag(p, p + 16);
        s = wmma_bf16(qf[c], kf, s);
      }
      #pragma unroll
      for (int r = 0; r < 8; ++r) {
        int qr = q0 + r + hlf * 8;
        float val = s[r] * 0.125f;         // 1/sqrt(64)
        sb[tt][r] = (key <= qr) ? val : -1e30f;
      }
    }

    // Online softmax: row reductions across the 16-lane halves
    float resc[8];
    #pragma unroll
    for (int r = 0; r < 8; ++r) {
      float mx = fmaxf(fmaxf(sb[0][r], sb[1][r]), fmaxf(sb[2][r], sb[3][r]));
      mx = fmaxf(mx, __shfl_xor(mx, 1, 32));
      mx = fmaxf(mx, __shfl_xor(mx, 2, 32));
      mx = fmaxf(mx, __shfl_xor(mx, 4, 32));
      mx = fmaxf(mx, __shfl_xor(mx, 8, 32));
      float mn = fmaxf(mrow[r], mx);
      resc[r] = __expf(mrow[r] - mn);
      mrow[r] = mn;
      float rs = 0.f;
      #pragma unroll
      for (int tt = 0; tt < 4; ++tt) {
        float p = __expf(sb[tt][r] - mn);
        sb[tt][r] = p;
        rs += p;
      }
      rs += __shfl_xor(rs, 1, 32);
      rs += __shfl_xor(rs, 2, 32);
      rs += __shfl_xor(rs, 4, 32);
      rs += __shfl_xor(rs, 8, 32);
      lrow[r] = lrow[r] * resc[r] + rs;
    }

    // P (C-layout) -> wave-private LDS (row-major) to re-load in A-layout
    __bf16* pp = lds_p[wave];
    #pragma unroll
    for (int tt = 0; tt < 4; ++tt)
      #pragma unroll
      for (int r = 0; r < 8; ++r)
        pp[(r + hlf * 8) * 72 + tt * 16 + l15] = (__bf16)sb[tt][r];

    // Rescale running O
    #pragma unroll
    for (int n = 0; n < 4; ++n)
      #pragma unroll
      for (int r = 0; r < 8; ++r)
        o[n][r] *= resc[r];

    // O += P(16x64) @ V(64x64)
    #pragma unroll
    for (int c = 0; c < 2; ++c) {
      const __bf16* ap = pp + l15 * 72 + c * 32 + hlf * 8;
      v16bf pf = make_frag(ap, ap + 16);
      #pragma unroll
      for (int n = 0; n < 4; ++n) {
        const __bf16* vp = lds_vt + (n * 16 + l15) * 72 + c * 32 + hlf * 8;
        v16bf vf = make_frag(vp, vp + 16);
        o[n] = wmma_bf16(pf, vf, o[n]);
      }
    }
  }

  // Normalize and store merged-head output [B,S,E] as bf16
  #pragma unroll
  for (int n = 0; n < 4; ++n) {
    int e = h * 64 + n * 16 + l15;
    #pragma unroll
    for (int r = 0; r < 8; ++r) {
      int srow = q0 + r + hlf * 8;
      size_t dst = ((size_t)b * S_ + srow) * E_ + e;
      attn_out[dst] = (__bf16)(o[n][r] / lrow[r]);
    }
  }
}

// ---------------------------------------------------------------------------
// Kernel 3: out = attn(bf16) @ W_proj + b_proj  (f32 output)
// Grid (32, 8), block 256, tile 128x128, K-step 64.
// ---------------------------------------------------------------------------
__global__ __launch_bounds__(256, 1)
void proj_gemm_kernel(const __bf16* __restrict__ Abf,
                      const float* __restrict__ W, const float* __restrict__ bias,
                      float* __restrict__ out) {
  __shared__ __bf16 lds_a[128 * 72];
  __shared__ __bf16 lds_b[128 * 72];

  const int t     = threadIdx.x;
  const int lane  = t & 31;
  const int wave  = t >> 5;
  const int waveM = wave >> 2;
  const int waveN = wave & 3;
  const int m0 = blockIdx.x * 128;
  const int n0 = blockIdx.y * 128;
  const int hlf = (lane >> 4) & 1;
  const int l15 = lane & 15;

  v8f acc[4][2];
  #pragma unroll
  for (int mi = 0; mi < 4; ++mi)
    #pragma unroll
    for (int ni = 0; ni < 2; ++ni)
      acc[mi][ni] = (v8f){0.f,0.f,0.f,0.f,0.f,0.f,0.f,0.f};

  for (int k = 0; k < E_; k += 64) {
    // Stage A (already bf16): straight 16B copies
    #pragma unroll
    for (int j = 0; j < 4; ++j) {
      int flat = t + j * 256;              // 1024 x 8-elt chunks
      int row  = flat >> 3;                // 8 chunks per row
      int col  = (flat & 7) * 8;
      const __bf16* gp = Abf + (size_t)(m0 + row) * E_ + k + col;
      v8bf v = *(const v8bf*)gp;
      if (k + 64 < E_) __builtin_prefetch(gp + 64, 0, 3);
      *(v8bf*)&lds_a[row * 72 + col] = v;
    }
    // Stage B transposed
    #pragma unroll
    for (int j = 0; j < 8; ++j) {
      int flat = t + j * 256;
      int krow = flat >> 5;
      int ncol = (flat & 31) * 4;
      const float* gp = W + (size_t)(k + krow) * E_ + n0 + ncol;
      const float4 v = *(const float4*)gp;
      if (k + 64 < E_) __builtin_prefetch(gp + (size_t)64 * E_, 0, 3);
      lds_b[(ncol + 0) * 72 + krow] = (__bf16)v.x;
      lds_b[(ncol + 1) * 72 + krow] = (__bf16)v.y;
      lds_b[(ncol + 2) * 72 + krow] = (__bf16)v.z;
      lds_b[(ncol + 3) * 72 + krow] = (__bf16)v.w;
    }
    __syncthreads();

    #pragma unroll
    for (int kc = 0; kc < 2; ++kc) {
      v16bf afrag[4];
      #pragma unroll
      for (int mi = 0; mi < 4; ++mi) {
        int row = waveM * 64 + mi * 16 + l15;
        const __bf16* p = lds_a + row * 72 + kc * 32 + hlf * 8;
        afrag[mi] = make_frag(p, p + 16);
      }
      v16bf bfrag[2];
      #pragma unroll
      for (int ni = 0; ni < 2; ++ni) {
        int n = waveN * 32 + ni * 16 + l15;
        const __bf16* p = lds_b + n * 72 + kc * 32 + hlf * 8;
        bfrag[ni] = make_frag(p, p + 16);
      }
      #pragma unroll
      for (int mi = 0; mi < 4; ++mi)
        #pragma unroll
        for (int ni = 0; ni < 2; ++ni)
          acc[mi][ni] = wmma_bf16(afrag[mi], bfrag[ni], acc[mi][ni]);
    }
    __syncthreads();
  }

  #pragma unroll
  for (int mi = 0; mi < 4; ++mi) {
    #pragma unroll
    for (int ni = 0; ni < 2; ++ni) {
      int n = n0 + waveN * 32 + ni * 16 + l15;
      float bv = bias[n];
      #pragma unroll
      for (int r = 0; r < 8; ++r) {
        int m = m0 + waveM * 64 + mi * 16 + r + hlf * 8;
        out[(size_t)m * E_ + n] = acc[mi][ni][r] + bv;
      }
    }
  }
}

extern "C" void kernel_launch(void* const* d_in, const int* in_sizes, int n_in,
                              void* d_out, int out_size, void* d_ws, size_t ws_size,
                              hipStream_t stream) {
  (void)in_sizes; (void)n_in; (void)out_size; (void)ws_size;
  const float* hs     = (const float*)d_in[0];
  const float* w_attn = (const float*)d_in[1];
  const float* b_attn = (const float*)d_in[2];
  const float* w_proj = (const float*)d_in[3];
  const float* b_proj = (const float*)d_in[4];
  float* out = (float*)d_out;

  // Workspace layout (bf16): QKV [3][B][H][S][HD] then attn output [B*S][E]
  __bf16* qkv  = (__bf16*)d_ws;                          // 25.2 MB
  __bf16* attn = qkv + (size_t)3 * B_ * H_ * S_ * HD_;   // +8.4 MB

  qkv_gemm_kernel<<<dim3(MTOT / 128, N3E / 128), 256, 0, stream>>>(
      hs, w_attn, b_attn, qkv);
  attn_kernel<<<dim3(S_ / 64, B_ * H_), 128, 0, stream>>>(qkv, attn);
  proj_gemm_kernel<<<dim3(MTOT / 128, E_ / 128), 256, 0, stream>>>(
      attn, w_proj, b_proj, out);
}
